// BlockwiseResonatorModel_82360292868778
// MI455X (gfx1250) — compile-verified, matrix operand order checked by reading, not run on hardware
//
#include <hip/hip_runtime.h>
#include <hip/hip_bf16.h>
#include <cstdint>
#include <cstddef>

// ---------------------------------------------------------------------------
// BlockwiseResonatorModel for MI455X (gfx1250, wave32, WMMA 16x16x32 f16)
//
//   [per head] embed GEMM (WMMA) -> 4x transposed-conv levels (WMMA, even/odd
//   phases as 2-tap GEMMs) -> final conv (N=1 heads: VALU dot; N=257 head:
//   3-tap WMMA GEMM with sigmoid*0.5 epilogue)
//   impulse: square -> linear resize 128->32768 -> *noise
//   rFFT-512 per frame (LDS radix-2) -> complex recurrence over frames ->
//   irFFT-512 -> overlap-add fused with the one-hot-delta shift into d_out.
//
//   GEMM blocks: 8 waves / 256 threads; each wave owns two 16-row tiles
//   (32x16 output) so every LDS B-fragment feeds two WMMAs. The W panel
//   (NTAPS x 16 cols x K=128, <=12KB) is staged into LDS once per block via
//   the Tensor Data Mover (tensor_load_to_lds + s_wait_tensorcnt), with a
//   cooperative-load fallback when the builtin is unavailable. NTAPS/MODE are
//   template params so the k-loop fully unrolls: loads batch into clauses,
//   WMMAs issue back-to-back.
// ---------------------------------------------------------------------------

typedef __attribute__((ext_vector_type(16))) _Float16 v16h;
typedef __attribute__((ext_vector_type(8)))  _Float16 h8;
typedef __attribute__((ext_vector_type(8)))  float    v8f;
typedef __attribute__((ext_vector_type(4)))  unsigned int u32x4;
typedef __attribute__((ext_vector_type(8)))  int          i32x8;
typedef __attribute__((ext_vector_type(4)))  int          i32x4;

#ifndef __has_builtin
#define __has_builtin(x) 0
#endif
#if __has_builtin(__builtin_amdgcn_tensor_load_to_lds)
#define HAVE_TDM 1
#else
#define HAVE_TDM 0
#endif

#define MODE_PLAIN 0
#define MODE_EMBED 1
#define MODE_LEAKY 2
#define MODE_MAG   3

#define NB    256      // batch
#define NCH   128      // channels (= GEMM K)
#define NFR   128      // frames
#define NSAMP 32768
#define STEPL 256
#define WINL  512
#define NCBIN 257

__device__ __forceinline__ v16h load_frag16(const _Float16* p, bool valid) {
  v16h a;
  if (valid) {
    h8 lo = *(const h8*)(p);
    h8 hi = *(const h8*)(p + 16);
#pragma unroll
    for (int i = 0; i < 8; ++i) { a[i] = lo[i]; a[i + 8] = hi[i]; }
  } else {
#pragma unroll
    for (int i = 0; i < 16; ++i) a[i] = (_Float16)0.0f;
  }
  return a;
}

#if HAVE_TDM
// TDM: copy a [rows<=16] x [128 f16] panel (row stride 128 elems) into LDS.
// D# group0: count=1 | lds_addr | 57b global addr | type=2 ("image").
// D# group1: data_size=1 (2B); tensor_dim0=128, tensor_dim1=rows_valid (OOB
// rows read as zero -> hardware zero-fill of the ragged last N-tile);
// tile_dim0=128, tile_dim1=16; tensor_dim0_stride=128.
__device__ __forceinline__ void tdm_load_w_tile(const _Float16* gsrc,
                                                unsigned lds_byte_addr,
                                                int rows_valid) {
  unsigned long long ga = (unsigned long long)(uintptr_t)gsrc;
  u32x4 g0;
  g0[0] = 1u;                                           // count=1 user D#
  g0[1] = lds_byte_addr;                                // lds_addr
  g0[2] = (unsigned)(ga & 0xffffffffu);                 // global_addr[31:0]
  g0[3] = (unsigned)((ga >> 32) & 0x01ffffffu) | (2u << 30);  // [56:32] | type=2
  i32x8 g1;
  g1[0] = (1 << 16);                   // data_size=1 (2 bytes), wg_mask=0
  g1[1] = (128 << 16);                 // tensor_dim0[15:0]=128 (bits 63:48)
  g1[2] = ((rows_valid & 0xffff) << 16); // tensor_dim1[15:0] (bits 95:80)
  g1[3] = (128 << 16);                 // tile_dim0=128 (bits 127:112)
  g1[4] = 16;                          // tile_dim1=16, tile_dim2=0
  g1[5] = 128;                         // tensor_dim0_stride[31:0]=128
  g1[6] = 0;                           // stride hi bits / dim1_stride
  g1[7] = 0;
  i32x4 z4 = {0, 0, 0, 0};
#if __clang_major__ >= 23
  i32x8 z8 = {0, 0, 0, 0, 0, 0, 0, 0};
  __builtin_amdgcn_tensor_load_to_lds(g0, g1, z4, z4, z8, 0);
#else
  __builtin_amdgcn_tensor_load_to_lds(g0, g1, z4, z4, 0);
#endif
}
#endif

template <int MODE>
__device__ __forceinline__ void gemm_epilogue(
    const v8f& acc, int rt, int lcol, bool hi, int totalRows, int Lin, int N,
    int ncol_base, const float* __restrict__ bias,
    float* __restrict__ outF, _Float16* __restrict__ outH,
    int Lout, int tmul, int tadd)
{
#pragma unroll
  for (int j = 0; j < 8; ++j) {
    const int m = rt * 16 + j + (hi ? 8 : 0);
    const int n = ncol_base + lcol;
    if (m >= totalRows || n >= N) continue;
    float v = acc[j] + (bias ? bias[n] : 0.0f);
    const int b = m / Lin;
    const int t = m - b * Lin;
    if (MODE == MODE_EMBED) {
      // (z@W+b).reshape(B, C, 8) consumed as [b][time=n%8][chan=n/8]
      outH[((size_t)(b * 8 + (n & 7))) * NCH + (n >> 3)] = (_Float16)v;
    } else if (MODE == MODE_LEAKY) {
      v = v > 0.0f ? v : 0.2f * v;
      const int trow = t * tmul + tadd;         // interleave even/odd phases
      outH[((size_t)(b * Lout + trow)) * NCH + n] = (_Float16)v;
    } else if (MODE == MODE_MAG) {
      v = 0.5f / (1.0f + __expf(-v));
      outF[((size_t)(b * NFR + t)) * NCBIN + n] = v;  // mag[b][frame][bin]
    } else {
      outF[((size_t)m) * N + n] = v;
    }
  }
}

// Generic multi-tap WMMA GEMM over channel dim (K = 128 fixed).
// A: [Bn*Lin][128] f16 row-major.  W*: [N][128] f16 (output col n holds
// K-contiguous weights).  out row r = (b = r/Lin, t = r%Lin); tap tp uses
// A row (b, t+sh), zero outside [0, Lin).  Each wave computes 32x16 output.
template <int NTAPS, int MODE>
__global__ void __launch_bounds__(256) wmma_gemm_kernel(
    const _Float16* __restrict__ A,
    const _Float16* __restrict__ W0,
    const _Float16* __restrict__ W1,
    const _Float16* __restrict__ W2,
    const float*    __restrict__ bias,
    float*          __restrict__ outF,
    _Float16*       __restrict__ outH,
    int Bn, int Lin, int N,
    int s0, int s1, int s2,
    int Lout, int tmul, int tadd)
{
  __shared__ _Float16 wlds[NTAPS * 16 * NCH];   // NTAPS x 16 cols x K (<=12KB)

  const int lane = threadIdx.x & 31;
  const int wave = threadIdx.x >> 5;            // 8 waves: two 16-row tiles each
  const int nt   = blockIdx.y;
  const int totalRows = Bn * Lin;
  const int ncol_base = nt * 16;
  int rows_valid = N - ncol_base; if (rows_valid > 16) rows_valid = 16;

  // ---- stage W panel (all taps) into LDS, once per block ----
#if HAVE_TDM
  if (wave == 0) {
    unsigned ldsbase = (unsigned)(uintptr_t)(void*)wlds;
#pragma unroll
    for (int tp = 0; tp < NTAPS; ++tp) {
      const _Float16* Wp = (tp == 0) ? W0 : ((tp == 1) ? W1 : W2);
      tdm_load_w_tile(Wp + (size_t)ncol_base * NCH,
                      ldsbase + (unsigned)(tp * 16 * NCH * 2), rows_valid);
    }
    __builtin_amdgcn_s_wait_tensorcnt(0);
  }
  __syncthreads();
#else
  for (int e = threadIdx.x; e < NTAPS * 16 * NCH; e += 256) {
    int tp  = e / (16 * NCH);
    int r   = e - tp * (16 * NCH);
    int row = r >> 7;
    const _Float16* Wp = (tp == 0) ? W0 : ((tp == 1) ? W1 : W2);
    wlds[e] = (row < rows_valid)
                  ? Wp[(size_t)(ncol_base + row) * NCH + (r & 127)]
                  : (_Float16)0.0f;
  }
  __syncthreads();
#endif

  const int rt0 = blockIdx.x * 16 + wave * 2;   // two consecutive row tiles
  const bool hi = lane >= 16;
  const int khalf = hi ? 8 : 0;
  const int lcol = lane & 15;

  const int mrow0 = rt0 * 16 + lcol;
  const int mrow1 = mrow0 + 16;
  const bool m0_ok = mrow0 < totalRows;
  const bool m1_ok = mrow1 < totalRows;
  const int b_m0 = m0_ok ? (mrow0 / Lin) : 0;
  const int t_m0 = mrow0 - b_m0 * Lin;
  const int b_m1 = m1_ok ? (mrow1 / Lin) : 0;
  const int t_m1 = mrow1 - b_m1 * Lin;

  v8f acc0 = {}, acc1 = {};
#pragma unroll
  for (int ks = 0; ks < 4; ++ks) {
    const int kofs = ks * 32 + khalf;
    v16h af0[NTAPS], af1[NTAPS], bf[NTAPS];
#pragma unroll
    for (int tp = 0; tp < NTAPS; ++tp) {
      const int sh = (tp == 0) ? s0 : ((tp == 1) ? s1 : s2);
      {
        const int t2 = t_m0 + sh;
        const bool av = m0_ok && (t2 >= 0) && (t2 < Lin);
        const _Float16* ap = A + ((size_t)(b_m0 * Lin + t2)) * NCH + kofs;
        if (av && ks == 0) __builtin_prefetch(ap + 32, 0, 1);  // global_prefetch_b8
        af0[tp] = load_frag16(ap, av);
      }
      {
        const int t2 = t_m1 + sh;
        const bool av = m1_ok && (t2 >= 0) && (t2 < Lin);
        af1[tp] = load_frag16(A + ((size_t)(b_m1 * Lin + t2)) * NCH + kofs, av);
      }
      const _Float16* wp = &wlds[tp * 16 * NCH + lcol * NCH + kofs];
      h8 lo = *(const h8*)(wp);
      h8 hh = *(const h8*)(wp + 16);
#pragma unroll
      for (int i = 0; i < 8; ++i) { bf[tp][i] = lo[i]; bf[tp][i + 8] = hh[i]; }
    }
#pragma unroll
    for (int tp = 0; tp < NTAPS; ++tp) {
      acc0 = __builtin_amdgcn_wmma_f32_16x16x32_f16(
          false, af0[tp], false, bf[tp], (short)0, acc0, false, false);
      acc1 = __builtin_amdgcn_wmma_f32_16x16x32_f16(
          false, af1[tp], false, bf[tp], (short)0, acc1, false, false);
    }
  }

  gemm_epilogue<MODE>(acc0, rt0,     lcol, hi, totalRows, Lin, N, ncol_base,
                      bias, outF, outH, Lout, tmul, tadd);
  gemm_epilogue<MODE>(acc1, rt0 + 1, lcol, hi, totalRows, Lin, N, ncol_base,
                      bias, outF, outH, Lout, tmul, tadd);
}

// ---------------- weight repack (f32 -> f16, GEMM layout [tap][N][K]) -------
__global__ void cvt_f16_kernel(const float* __restrict__ s, _Float16* __restrict__ d, int n) {
  int i = blockIdx.x * blockDim.x + threadIdx.x;
  if (i < n) d[i] = (_Float16)s[i];
}
__global__ void repack_embed_kernel(const float* __restrict__ w, _Float16* __restrict__ wt) {
  int i = blockIdx.x * blockDim.x + threadIdx.x;      // n*128 + k, n < 1024
  if (i >= 1024 * NCH) return;
  int n = i >> 7, k = i & 127;
  wt[i] = (_Float16)w[(size_t)k * 1024 + n];          // transpose (K,N)->(N,K)
}
__global__ void repack_up_kernel(const float* __restrict__ w, int lvl, _Float16* __restrict__ wt) {
  int i = blockIdx.x * blockDim.x + threadIdx.x;      // tap*16384 + n*128 + k
  if (i >= 4 * NCH * NCH) return;
  int tap = i >> 14, n = (i >> 7) & 127, k = i & 127;
  wt[i] = (_Float16)w[(((size_t)(lvl * NCH + n)) * NCH + k) * 4 + tap];
}
__global__ void repack_final_kernel(const float* __restrict__ w, _Float16* __restrict__ wt) {
  int i = blockIdx.x * blockDim.x + threadIdx.x;      // tap*(257*128) + n*128 + k
  if (i >= 3 * NCBIN * NCH) return;
  int tap = i / (NCBIN * NCH);
  int r = i - tap * (NCBIN * NCH);
  int n = r >> 7, k = r & 127;
  wt[i] = (_Float16)w[((size_t)n * NCH + k) * 3 + tap];
}

// ---------------- out_ch==1 final conv (loc / impulse heads) ---------------
__global__ void final1_kernel(const _Float16* __restrict__ H,   // [B][128][128]
                              const float* __restrict__ fw,     // [1][128][3]
                              const float* __restrict__ fb,
                              float* __restrict__ out, int do_square)
{
  int idx = blockIdx.x * blockDim.x + threadIdx.x;    // b*128 + t
  if (idx >= NB * NFR) return;
  int b = idx >> 7, t = idx & 127;
  float acc = fb[0];
  for (int tp = 0; tp < 3; ++tp) {
    int t2 = t + tp - 1;
    if (t2 < 0 || t2 >= NFR) continue;
    const _Float16* hp = H + ((size_t)(b * NFR + t2)) * NCH;
    float s = 0.0f;
    for (int k = 0; k < NCH; ++k) s += (float)hp[k] * fw[k * 3 + tp];
    acc += s;
  }
  out[idx] = do_square ? acc * acc : acc;
}

// ---------------- impulse: linear resize 128->32768, * noise ---------------
__global__ void impulse_resize_kernel(const float* __restrict__ imp128,
                                      const float* __restrict__ noise,
                                      float* __restrict__ impfull)
{
  int idx = blockIdx.x * blockDim.x + threadIdx.x;
  if (idx >= NB * NSAMP) return;
  int b = idx >> 15, i = idx & (NSAMP - 1);
  float pos = (i + 0.5f) * (1.0f / 256.0f) - 0.5f;    // half-pixel centers
  float fl = floorf(pos);
  int i0 = (int)fl;
  float w = pos - fl;
  int ia = i0 < 0 ? 0 : (i0 > 127 ? 127 : i0);
  int ib = i0 + 1 < 0 ? 0 : (i0 + 1 > 127 ? 127 : i0 + 1);
  float v = imp128[b * NFR + ia] * (1.0f - w) + imp128[b * NFR + ib] * w;
  impfull[(size_t)b * NSAMP + i] = v * noise[i];
}

__device__ __forceinline__ int brev8(int v) {
  int r = 0;
#pragma unroll
  for (int i = 0; i < 8; ++i) r = (r << 1) | ((v >> i) & 1);
  return r;
}

// ---------------- rFFT-512 (hamming windowed frame), ortho -----------------
__global__ void rfft512_kernel(const float* __restrict__ impfull,
                               float* __restrict__ fre, float* __restrict__ fim)
{
  __shared__ float zr[256], zi[256];
  const int frame = blockIdx.x;                 // b*128 + f
  const int b = frame >> 7, f = frame & 127;
  const int j = threadIdx.x;                    // 0..127
  const float PI2 = 6.28318530717958647692f;

  for (int mm = j; mm < 256; mm += 128) {       // even/odd pack z[m]
    int s0 = f * STEPL + 2 * mm;
    int s1 = s0 + 1;
    float x0 = (s0 < NSAMP) ? impfull[(size_t)b * NSAMP + s0] : 0.0f;
    float x1 = (s1 < NSAMP) ? impfull[(size_t)b * NSAMP + s1] : 0.0f;
    float h0 = 0.54f - 0.46f * __cosf(PI2 * (float)(2 * mm) / 512.0f);
    float h1 = 0.54f - 0.46f * __cosf(PI2 * (float)(2 * mm + 1) / 512.0f);
    int rm = brev8(mm);
    zr[rm] = x0 * h0; zi[rm] = x1 * h1;
  }
  __syncthreads();
  for (int s = 0; s < 8; ++s) {
    int span = 1 << s;
    int i1 = ((j >> s) << (s + 1)) | (j & (span - 1));
    int i2 = i1 + span;
    float ang = -PI2 * (float)(j & (span - 1)) / (float)(2 * span);
    float wr = __cosf(ang), wi = __sinf(ang);
    float ar = zr[i1], ai = zi[i1];
    float br = zr[i2], bi = zi[i2];
    float tr = wr * br - wi * bi, ti = wr * bi + wi * br;
    zr[i1] = ar + tr; zi[i1] = ai + ti;
    zr[i2] = ar - tr; zi[i2] = ai - ti;
    __syncthreads();
  }
  const float SC = 0.0441941738241592f;         // 1/sqrt(512)
  size_t ob = (size_t)frame * NCBIN;
  if (j == 0) {
    float r0 = zr[0], i0 = zi[0];
    fre[ob + 0]   = (r0 + i0) * SC; fim[ob + 0]   = 0.0f;
    fre[ob + 256] = (r0 - i0) * SC; fim[ob + 256] = 0.0f;
    fre[ob + 128] = zr[128] * SC;   fim[ob + 128] = -zi[128] * SC;
  } else {
    int k = j;                                  // 1..127
    float ar = zr[k], ai = zi[k];
    float br = zr[256 - k], bi = zi[256 - k];
    float Er = 0.5f * (ar + br), Ei = 0.5f * (ai - bi);
    float Dr = 0.5f * (ar - br), Di = 0.5f * (ai + bi);
    float Or = Di, Oi = -Dr;                    // D / i
    float ang = -PI2 * (float)k / 512.0f;
    float wr = __cosf(ang), wi = __sinf(ang);
    float WOr = wr * Or - wi * Oi, WOi = wr * Oi + wi * Or;
    fre[ob + k] = (Er + WOr) * SC;       fim[ob + k] = (Ei + WOi) * SC;
    fre[ob + (256 - k)] = (Er - WOr) * SC; fim[ob + (256 - k)] = -(Ei - WOi) * SC;
  }
}

// ---------------- per-(b,bin) frame recurrence: s = f + s*m ----------------
__global__ void recurrence_kernel(float* __restrict__ fre, float* __restrict__ fim,
                                  const float* __restrict__ mag)
{
  int idx = blockIdx.x * blockDim.x + threadIdx.x;    // b*257 + bin
  if (idx >= NB * NCBIN) return;
  int b = idx / NCBIN, bin = idx - b * NCBIN;
  float sr = 0.0f, si = 0.0f;
  for (int f = 0; f < NFR; ++f) {
    size_t o = ((size_t)(b * NFR + f)) * NCBIN + bin;
    float m = mag[o];
    sr = fre[o] + sr * m;
    si = fim[o] + si * m;
    fre[o] = sr; fim[o] = si;
  }
}

// ---------------- irFFT-512, ortho -----------------------------------------
__global__ void irfft512_kernel(const float* __restrict__ fre,
                                const float* __restrict__ fim,
                                float* __restrict__ frames)
{
  __shared__ float zr[256], zi[256];
  const int frame = blockIdx.x;
  const int j = threadIdx.x;                    // 0..127
  size_t ib = (size_t)frame * NCBIN;
  const float PI2 = 6.28318530717958647692f;

  for (int k = j; k < 256; k += 128) {
    float Xr = fre[ib + k], Xi = fim[ib + k];
    int k2 = 256 - k;
    float Yr = fre[ib + k2], Yi = fim[ib + k2];
    float Er = 0.5f * (Xr + Yr), Ei = 0.5f * (Xi - Yi);
    float Dr = 0.5f * (Xr - Yr), Di = 0.5f * (Xi + Yi);
    float ang = PI2 * (float)k / 512.0f;        // conj(W) untwist
    float wr = __cosf(ang), wi = __sinf(ang);
    float Or = wr * Dr - wi * Di, Oi = wr * Di + wi * Dr;
    float Zr = Er - Oi, Zi = Ei + Or;           // Z = E + i*O
    int rm = brev8(k);
    zr[rm] = Zr; zi[rm] = Zi;
  }
  __syncthreads();
  for (int s = 0; s < 8; ++s) {
    int span = 1 << s;
    int i1 = ((j >> s) << (s + 1)) | (j & (span - 1));
    int i2 = i1 + span;
    float ang = PI2 * (float)(j & (span - 1)) / (float)(2 * span);   // inverse
    float wr = __cosf(ang), wi = __sinf(ang);
    float ar = zr[i1], ai = zi[i1];
    float br = zr[i2], bi = zi[i2];
    float tr = wr * br - wi * bi, ti = wr * bi + wi * br;
    zr[i1] = ar + tr; zi[i1] = ai + ti;
    zr[i2] = ar - tr; zi[i2] = ai - ti;
    __syncthreads();
  }
  const float ISC = 0.0883883476483184f;        // sqrt(512)/256
  for (int m = j; m < 256; m += 128) {
    frames[(size_t)frame * WINL + 2 * m]     = zr[m] * ISC;
    frames[(size_t)frame * WINL + 2 * m + 1] = zi[m] * ISC;
  }
}

// ---------------- argmax(logits + gumbel) -> sample delay ------------------
__global__ void argmax_kernel(const float* __restrict__ logits,
                              const float* __restrict__ gumbel,
                              int* __restrict__ dstar)
{
  __shared__ float sv[128]; __shared__ int six[128];
  int b = blockIdx.x, t = threadIdx.x;
  sv[t] = logits[b * NFR + t] + gumbel[b * NFR + t];
  six[t] = t;
  __syncthreads();
  for (int s = 64; s > 0; s >>= 1) {
    if (t < s) {
      if (sv[t + s] > sv[t] || (sv[t + s] == sv[t] && six[t + s] < six[t])) {
        sv[t] = sv[t + s]; six[t] = six[t + s];
      }
    }
    __syncthreads();
  }
  if (t == 0) dstar[b] = six[0] * (NSAMP / NCH);   // delta position
}

// ---------------- overlap-add fused with one-hot-delta shift ---------------
__global__ void fill_zero_kernel(float* __restrict__ p, int n) {
  int i = blockIdx.x * blockDim.x + threadIdx.x;
  if (i < n) p[i] = 0.0f;
}
__global__ void ola_shift_kernel(const float* __restrict__ frames,
                                 const int* __restrict__ dstar,
                                 float* __restrict__ out)
{
  int frame = blockIdx.x;                       // b*128 + f
  int b = frame >> 7, f = frame & 127;
  int j = threadIdx.x;                          // 0..255
  float v = frames[(size_t)frame * WINL + j];
  if (f > 0) v += frames[(size_t)(frame - 1) * WINL + STEPL + j];
  int n = f * STEPL + j + dstar[b];
  if (n < NSAMP) out[(size_t)b * NSAMP + n] = v;
}

// ===========================================================================
extern "C" void kernel_launch(void* const* d_in, const int* in_sizes, int n_in,
                              void* d_out, int out_size, void* d_ws, size_t ws_size,
                              hipStream_t stream)
{
  // ---- resolve input layout (insertion order vs jax-sorted vs one blob) ----
  const float *x, *gumbel, *noise;
  const float *p_ew[3], *p_eb[3], *p_uw[3], *p_ub[3], *p_fw[3], *p_fb[3]; // loc, imp, transfer
  if (n_in >= 21) {
    bool ins = (in_sizes[1] == 131072);   // to_loc.embed_w vs noise
    if (ins) {
      x = (const float*)d_in[0]; gumbel = (const float*)d_in[19]; noise = (const float*)d_in[20];
      int base[3] = {1, 7, 13};           // to_loc, to_impulse, to_transfer
      for (int h = 0; h < 3; ++h) {
        p_ew[h] = (const float*)d_in[base[h] + 0];
        p_eb[h] = (const float*)d_in[base[h] + 1];
        p_uw[h] = (const float*)d_in[base[h] + 2];
        p_ub[h] = (const float*)d_in[base[h] + 3];
        p_fw[h] = (const float*)d_in[base[h] + 4];
        p_fb[h] = (const float*)d_in[base[h] + 5];
      }
    } else {                               // sorted: gumbel, noise, params{imp,loc,transfer}{eb,ew,fb,fw,ub,uw}, x
      gumbel = (const float*)d_in[0]; noise = (const float*)d_in[1]; x = (const float*)d_in[20];
      int base[3] = {8, 2, 14};           // loc, impulse, transfer bases
      for (int h = 0; h < 3; ++h) {
        p_eb[h] = (const float*)d_in[base[h] + 0];
        p_ew[h] = (const float*)d_in[base[h] + 1];
        p_fb[h] = (const float*)d_in[base[h] + 2];
        p_fw[h] = (const float*)d_in[base[h] + 3];
        p_ub[h] = (const float*)d_in[base[h] + 4];
        p_uw[h] = (const float*)d_in[base[h] + 5];
      }
    }
  } else {
    // params flattened into one blob (insertion order, heads contiguous)
    x = (const float*)d_in[0];
    const float* blob = (const float*)d_in[1];
    gumbel = (const float*)d_in[2]; noise = (const float*)d_in[3];
    size_t off = 0;
    for (int h = 0; h < 3; ++h) {
      int outc = (h == 2) ? 514 : 1;
      p_ew[h] = blob + off; off += 128 * 1024;
      p_eb[h] = blob + off; off += 1024;
      p_uw[h] = blob + off; off += 4 * 128 * 128 * 4;
      p_ub[h] = blob + off; off += 4 * 128;
      p_fw[h] = blob + off; off += (size_t)outc * 128 * 3;
      p_fb[h] = blob + off; off += outc;
    }
  }

  // ---- workspace carve ----
  char* ws = (char*)d_ws;
  size_t off = 0;
  auto carve = [&](size_t bytes) -> void* {
    void* p = ws + off;
    off = (off + bytes + 255) & ~(size_t)255;
    return p;
  };
  _Float16* Wt      = (_Float16*)carve((size_t)1024 * NCH * sizeof(_Float16));  // max repack
  _Float16* xh      = (_Float16*)carve((size_t)NB * NCH * sizeof(_Float16));
  _Float16* Aping   = (_Float16*)carve((size_t)NB * NFR * NCH * sizeof(_Float16));
  _Float16* Apong   = (_Float16*)carve((size_t)NB * NFR * NCH * sizeof(_Float16));
  float*    logits  = (float*)carve((size_t)NB * NFR * sizeof(float));
  float*    imp128  = (float*)carve((size_t)NB * NFR * sizeof(float));
  float*    magB    = (float*)carve((size_t)NB * NFR * NCBIN * sizeof(float));
  float*    fre     = (float*)carve((size_t)NB * NFR * NCBIN * sizeof(float));
  float*    fim     = (float*)carve((size_t)NB * NFR * NCBIN * sizeof(float));
  float*    impfull = (float*)carve((size_t)NB * NSAMP * sizeof(float));
  float*    framesB = (float*)carve((size_t)NB * NFR * WINL * sizeof(float));
  int*      dstar   = (int*)carve((size_t)NB * sizeof(int));
  (void)ws_size;

  auto grid_for = [](int rows, int N) {
    return dim3((unsigned)((rows + 255) / 256), (unsigned)((N + 15) / 16));
  };

  cvt_f16_kernel<<<(NB * NCH + 255) / 256, 256, 0, stream>>>(x, xh, NB * NCH);

  for (int h = 0; h < 3; ++h) {   // 0=loc, 1=impulse, 2=transfer
    // embed: (256x128) @ (128x1024) -> reshape to (B, 8, 128) f16
    repack_embed_kernel<<<(1024 * NCH + 255) / 256, 256, 0, stream>>>(p_ew[h], Wt);
    wmma_gemm_kernel<1, MODE_EMBED><<<grid_for(NB * 1, 1024), 256, 0, stream>>>(
        xh, Wt, Wt, Wt, p_eb[h], nullptr, Aping,
        NB, 1, 1024, 0, 0, 0, 8, 1, 0);

    _Float16* A0 = Aping; _Float16* A1 = Apong;
    int L = 8;
    for (int lvl = 0; lvl < 4; ++lvl) {
      repack_up_kernel<<<(4 * NCH * NCH + 255) / 256, 256, 0, stream>>>(p_uw[h], lvl, Wt);
      const size_t blk = (size_t)NCH * NCH;
      // even outputs: x[t-1]*W0 + x[t]*W2 ; odd: x[t]*W1 + x[t+1]*W3
      wmma_gemm_kernel<2, MODE_LEAKY><<<grid_for(NB * L, NCH), 256, 0, stream>>>(
          A0, Wt + 0 * blk, Wt + 2 * blk, Wt, p_ub[h] + lvl * NCH, nullptr, A1,
          NB, L, NCH, -1, 0, 0, 2 * L, 2, 0);
      wmma_gemm_kernel<2, MODE_LEAKY><<<grid_for(NB * L, NCH), 256, 0, stream>>>(
          A0, Wt + 1 * blk, Wt + 3 * blk, Wt, p_ub[h] + lvl * NCH, nullptr, A1,
          NB, L, NCH, 0, 1, 0, 2 * L, 2, 1);
      _Float16* t = A0; A0 = A1; A1 = t;
      L *= 2;
    }
    // A0 now holds (B, 128, 128) f16 activations
    if (h == 0) {
      final1_kernel<<<(NB * NFR + 255) / 256, 256, 0, stream>>>(A0, p_fw[h], p_fb[h], logits, 0);
    } else if (h == 1) {
      final1_kernel<<<(NB * NFR + 255) / 256, 256, 0, stream>>>(A0, p_fw[h], p_fb[h], imp128, 1);
    } else {
      // only the first 257 of 514 output channels are consumed downstream
      repack_final_kernel<<<(3 * NCBIN * NCH + 255) / 256, 256, 0, stream>>>(p_fw[h], Wt);
      const size_t fblk = (size_t)NCBIN * NCH;
      wmma_gemm_kernel<3, MODE_MAG><<<grid_for(NB * NFR, NCBIN), 256, 0, stream>>>(
          A0, Wt + 0 * fblk, Wt + 1 * fblk, Wt + 2 * fblk, p_fb[h], magB, nullptr,
          NB, NFR, NCBIN, -1, 0, 1, NFR, 1, 0);
    }
  }

  impulse_resize_kernel<<<(NB * NSAMP + 255) / 256, 256, 0, stream>>>(imp128, noise, impfull);
  rfft512_kernel<<<NB * NFR, 128, 0, stream>>>(impfull, fre, fim);
  recurrence_kernel<<<(NB * NCBIN + 255) / 256, 256, 0, stream>>>(fre, fim, magB);
  irfft512_kernel<<<NB * NFR, 128, 0, stream>>>(fre, fim, framesB);

  fill_zero_kernel<<<(NB * NSAMP + 255) / 256, 256, 0, stream>>>((float*)d_out, NB * NSAMP);
  argmax_kernel<<<NB, 128, 0, stream>>>(logits, gumbel, dstar);
  ola_shift_kernel<<<NB * NFR, STEPL, 0, stream>>>(framesB, dstar, (float*)d_out);
  (void)out_size; (void)n_in;
}
